// CRF_33457795236422
// MI455X (gfx1250) — compile-verified, matrix operand order checked by reading, not run on hardware
//
#include <hip/hip_runtime.h>
#include <stdint.h>
#include <stddef.h>

// Problem dims (fixed by the reference).
#define B_ 256
#define T_ 512
#define C_ 128

// Async staging ring for potentials.
#define RING 4
#define PREF 3  // RING-1 steps of lookahead

// d_out layout (floats): [viterbi B*T | inputs B*T*C | seq_len B | transitions C*C]
#define OFF_INP (B_ * T_)
#define OFF_LEN (OFF_INP + B_ * T_ * C_)
#define OFF_TRN (OFF_LEN + B_)

typedef int v8i __attribute__((ext_vector_type(8)));

struct __align__(16) Smem {
  float state[2][C_];            // ping-pong DP state (float4-aligned)
  float pot[RING][C_];           // async-staged potentials ring
  int   btag;                    // last_tag broadcast
  unsigned char bp[T_ - 1][C_];  // backpointers, u8 per tag per step
};

// CDNA5 async global->LDS copy (ASYNCcnt-tracked). LDS dest offset is a
// per-lane VGPR byte offset relative to the wave's LDS base; Smem is the only
// shared object so its base offset is 0 and offsetof() gives the LDS address.
__device__ __forceinline__ void async_ld_b32(unsigned lds_off, const float* g) {
  asm volatile("global_load_async_to_lds_b32 %0, %1, off"
               :: "v"(lds_off), "v"((unsigned long long)(uintptr_t)g)
               : "memory");
}
#define ASYNC_WAIT(N) asm volatile("s_wait_asynccnt " #N ::: "memory")

// ---------------------------------------------------------------------------
// seq_len = mask @ ones via V_WMMA_I32_16X16X64_IU8.
// One wave per 16 batches. A = 16x64 u8 mask slice (8 chunks over T=512),
// B = all-ones bytes, C accumulates; D[m][n] == seq_len[m] for every column n.
// 8-bit A layout: lane l holds M=l&15; lanes>=16 carry K+8; VGPR v covers
// K-bytes { (v&1)*4 + ((v>>1)&1)*16 + (v>>2)*32 } + beta.
// ---------------------------------------------------------------------------
__global__ __launch_bounds__(32) void seqlen_wmma_kernel(
    const int* __restrict__ mask,  // [B,T] 0/1
    int*       __restrict__ lens,  // [B] int (workspace)
    float*     __restrict__ out) { // seq_len floats at OFF_LEN
  const int l = threadIdx.x;                       // lane 0..31
  const int mrow = (blockIdx.x << 4) + (l & 15);   // batch index for A row
  const int khi = (l >> 4) * 8;                    // lanes 16..31: K+8
  const int* mrowp = mask + mrow * T_;

  v8i c = {0, 0, 0, 0, 0, 0, 0, 0};
  const v8i bones = {0x01010101, 0x01010101, 0x01010101, 0x01010101,
                     0x01010101, 0x01010101, 0x01010101, 0x01010101};
#pragma unroll
  for (int kk = 0; kk < T_ / 64; ++kk) {
    v8i a;
#pragma unroll
    for (int v = 0; v < 8; ++v) {
      const int k0 = ((v & 1) * 4) + (((v >> 1) & 1) * 16) + ((v >> 2) * 32) + khi + kk * 64;
      int p = 0;
#pragma unroll
      for (int bb = 0; bb < 4; ++bb)
        p |= (mrowp[k0 + bb] != 0 ? 1 : 0) << (8 * bb);
      a[v] = p;
    }
    c = __builtin_amdgcn_wmma_i32_16x16x64_iu8(false, a, false, bones, c,
                                               false, false);
  }
  // Column 0 of D: lane 0 -> tile rows 0..7, lane 16 -> tile rows 8..15.
  if ((l & 15) == 0) {
    const int rbase = (blockIdx.x << 4) + (l >> 4) * 8;
#pragma unroll
    for (int g = 0; g < 8; ++g) {
      lens[rbase + g] = c[g];
      out[OFF_LEN + rbase + g] = (float)c[g];
    }
  }
}

// ---------------------------------------------------------------------------
// Max-plus Viterbi DP: one block per batch, thread j owns tag j; trans[:,j]
// lives in 128 VGPRs; state ping-pongs in LDS; potentials stream in via
// async global->LDS with a 4-slot ring; backpointers + backtrack in LDS.
// ---------------------------------------------------------------------------
__global__ __launch_bounds__(C_) void crf_viterbi_kernel(
    const float* __restrict__ inp,    // [B,T,C]
    const float* __restrict__ trans,  // [C,C]
    const int*   __restrict__ lens,   // [B] seq_len (from WMMA kernel)
    float*       __restrict__ out) {
  __shared__ Smem sm;
  const int b = blockIdx.x;
  const int j = threadIdx.x;
  const size_t inBase = (size_t)b * T_ * C_;

  // ---- initial state = inputs[b, 0, :]
  sm.state[0][j] = inp[inBase + j];

  // ---- transition column trans[:, j] into 128 VGPRs (fully unrolled => SROA).
  float col[C_];
#pragma unroll
  for (int i = 0; i < C_; ++i) col[i] = trans[i * C_ + j];

  const int L = lens[b] - 1;  // DP steps t = 1..L; t >= seq_len are identity.
  __syncthreads();

  const unsigned potBase = (unsigned)offsetof(Smem, pot);

  // ---- async prefetch prologue: stage potentials for t = 1..min(PREF, L)
  for (int tp = 1; tp <= L && tp <= PREF; ++tp)
    async_ld_b32(potBase + (unsigned)((tp & (RING - 1)) * (C_ * 4)) + (unsigned)j * 4u,
                 inp + inBase + (size_t)tp * C_ + j);

  // ---- max-plus forward DP
  int cur = 0;
  for (int t = 1; t <= L; ++t) {
    const bool deep = (t + PREF) <= L;  // uniform per block
    if (deep)
      async_ld_b32(potBase + (unsigned)(((t + PREF) & (RING - 1)) * (C_ * 4)) + (unsigned)j * 4u,
                   inp + inBase + (size_t)(t + PREF) * C_ + j);

    // 4 independent max/argmax chains -> 4x dependency-chain ILP.
    float b0 = -__builtin_inff(), b1 = -__builtin_inff();
    float b2 = -__builtin_inff(), b3 = -__builtin_inff();
    int i0 = 0, i1 = 1, i2 = 2, i3 = 3;
#pragma unroll
    for (int i = 0; i < C_; i += 4) {
      const float4 sv = *(const float4*)&sm.state[cur][i];  // ds_load_b128 broadcast
      float x0 = sv.x + col[i + 0]; if (x0 > b0) { b0 = x0; i0 = i + 0; }
      float x1 = sv.y + col[i + 1]; if (x1 > b1) { b1 = x1; i1 = i + 1; }
      float x2 = sv.z + col[i + 2]; if (x2 > b2) { b2 = x2; i2 = i + 2; }
      float x3 = sv.w + col[i + 3]; if (x3 > b3) { b3 = x3; i3 = i + 3; }
    }
    // Merge with exact first-occurrence tie-break (smaller index on equal).
    float best = b0; int bidx = i0;
    if (b1 > best || (b1 == best && i1 < bidx)) { best = b1; bidx = i1; }
    if (b2 > best || (b2 == best && i2 < bidx)) { best = b2; bidx = i2; }
    if (b3 > best || (b3 == best && i3 < bidx)) { best = b3; bidx = i3; }

    // Wait for this step's staged potentials (this wave issued its own lanes).
    if (deep) { ASYNC_WAIT(3); } else { ASYNC_WAIT(0); }
    const float pot = sm.pot[t & (RING - 1)][j];

    sm.bp[t - 1][j] = (unsigned char)bidx;
    const int nxt = cur ^ 1;
    sm.state[nxt][j] = best + pot;
    cur = nxt;
    __syncthreads();  // single barrier per step (ping-pong state)
  }

  // ---- last_tag = first-occurrence argmax of final state.
  if (j == 0) {
    float bv = sm.state[cur][0];
    int bt = 0;
    for (int k = 1; k < C_; ++k) {
      const float v = sm.state[cur][k];
      if (v > bv) { bv = v; bt = k; }
    }
    sm.btag = bt;
  }
  __syncthreads();

  const int bt = sm.btag;
  float* vout = out + (size_t)b * T_;
  // t in [seq_len-1, T-1] all equal last_tag (identity bps) -> parallel fill.
  const int tail0 = (L < 0) ? 0 : L;
  for (int t = tail0 + j; t < T_; t += C_) vout[t] = (float)bt;
  // Serial backtrack over real backpointers (LDS-resident, ~seq_len hops).
  if (j == 0) {
    int tag = bt;
    for (int t = L - 1; t >= 0; --t) {
      tag = (int)sm.bp[t][tag];
      vout[t] = (float)tag;
    }
  }
}

// Vectorized echo of inputs / transitions into d_out (memory-bound, trivial).
__global__ void copy_f4_kernel(const float4* __restrict__ src, float4* __restrict__ dst, int n) {
  int i = blockIdx.x * blockDim.x + threadIdx.x;
  const int stride = gridDim.x * blockDim.x;
  for (; i < n; i += stride) dst[i] = src[i];
}

extern "C" void kernel_launch(void* const* d_in, const int* in_sizes, int n_in,
                              void* d_out, int out_size, void* d_ws, size_t ws_size,
                              hipStream_t stream) {
  (void)in_sizes; (void)n_in; (void)out_size; (void)ws_size;
  const float* inp   = (const float*)d_in[0];  // [B,T,C] f32
  const int*   mask  = (const int*)d_in[1];    // [B,T] bool->int
  const float* trans = (const float*)d_in[2];  // [C,C] f32
  float* out = (float*)d_out;
  int*   lens = (int*)d_ws;                    // [B] seq_len scratch

  // Echo inputs and transitions (tuple outputs 2 and 4).
  copy_f4_kernel<<<4096, 256, 0, stream>>>((const float4*)inp,
                                           (float4*)(out + OFF_INP),
                                           (B_ * T_ * C_) / 4);
  copy_f4_kernel<<<16, 256, 0, stream>>>((const float4*)trans,
                                         (float4*)(out + OFF_TRN),
                                         (C_ * C_) / 4);

  // seq_len via IU8 WMMA (tuple output 3 + int scratch for the DP kernel).
  seqlen_wmma_kernel<<<B_ / 16, 32, 0, stream>>>(mask, lens, out);

  // Viterbi decode (tuple output 1).
  crf_viterbi_kernel<<<B_, C_, 0, stream>>>(inp, trans, lens, out);
}
